// SparseAttention_16647293239593
// MI455X (gfx1250) — compile-verified, matrix operand order checked by reading, not run on hardware
//
#include <hip/hip_runtime.h>
#include <stdint.h>

#define T_SEQ 4096
#define DMODEL 512
#define NHEAD 8
#define DHEAD 64
#define WIN 64

typedef __attribute__((ext_vector_type(16))) __bf16 bf16x16;
typedef __attribute__((ext_vector_type(8)))  float  f32x8;

union FragAB {
    bf16x16 v;
    uint4   u[2];
};

static __device__ __forceinline__ int imin(int a, int b) { return a < b ? a : b; }
static __device__ __forceinline__ int imax(int a, int b) { return a > b ? a : b; }

// float -> bf16 bits, round-to-nearest-even
static __device__ __forceinline__ unsigned short f2bf(float f) {
    union { float f; uint32_t u; } c; c.f = f;
    uint32_t u = c.u;
    uint32_t r = u + 0x7FFFu + ((u >> 16) & 1u);
    return (unsigned short)(r >> 16);
}

// Intra-wave XOR shuffle via ds_swizzle_b32 (group-of-32 mode: and=0x1F, or=0, xor=MASK)
template <int MASK>
static __device__ __forceinline__ float swz_xor(float x) {
    return __int_as_float(
        __builtin_amdgcn_ds_swizzle(__float_as_int(x), (MASK << 10) | 0x1F));
}

// Async 16B copy global -> LDS (tracked by ASYNCcnt)
static __device__ __forceinline__ void async_cp16(uint32_t lds_addr, uint64_t gaddr) {
    asm volatile("global_load_async_to_lds_b128 %0, %1, off"
                 :: "v"(lds_addr), "v"(gaddr) : "memory");
}

__global__ void convert_f32_bf16(const float* __restrict__ in,
                                 unsigned short* __restrict__ out, int n) {
    int i = blockIdx.x * blockDim.x + threadIdx.x;
    int stride = gridDim.x * blockDim.x;
    for (; i < n; i += stride) out[i] = f2bf(in[i]);
}

// C[M=4096, N=512] = A[4096,512] @ W^T (W rows are output columns).
// MODE 0: bf16 row-major. MODE 1: bf16 transposed [N][T]. MODE 2: f32 row-major.
// Block = 128 threads = 4 waves; each wave owns a 16x64 tile. The block shares a
// 64x64 W k-slice (8KB), double-buffered in LDS via async global->LDS copies;
// each barrier pair covers 2 k-steps = 8 WMMAs per wave.
template <int MODE>
__global__ void __launch_bounds__(128)
gemm_x_wT(const unsigned short* __restrict__ A,
          const unsigned short* __restrict__ W,
          unsigned short* __restrict__ Obf,
          float* __restrict__ Of) {
    __shared__ __align__(16) unsigned short wbuf[2][64][64];

    const int tid  = threadIdx.x;
    const int lane = tid & 31;
    const int wave = tid >> 5;
    const int row  = lane & 15;   // M row of A-frag / N col of B-frag & C tile
    const int half = lane >> 4;   // K sub-block select / C row offset
    const int m0 = (blockIdx.x * 4 + wave) * 16;
    const int n0 = blockIdx.y * 64;
    const int koff = half * 8;

    // stage W[n0..n0+63][s*64..+63] into wbuf[s&1]; 4 x 16B per thread (8KB total)
    auto stage = [&](int s) {
#pragma unroll
        for (int i = 0; i < 4; ++i) {
            const int c  = tid * 4 + i;          // 0..511 chunks of 16B
            const int r  = c >> 3;               // W row within tile (8 chunks/row)
            const int cb = (c & 7) * 8;          // short offset within 64-short row
            const uint64_t g = (uint64_t)(const void*)(W + (size_t)(n0 + r) * DMODEL + s * 64 + cb);
            const uint32_t l = (uint32_t)(size_t)&wbuf[s & 1][r][cb];
            async_cp16(l, g);
        }
    };

    f32x8 acc[4] = {};
    const unsigned short* arow = A + (size_t)(m0 + row) * DMODEL;

    stage(0);
    for (int s = 0; s < 8; ++s) {
        const int buf = s & 1;
        if (s < 7) {
            stage(s + 1);
            asm volatile("s_wait_asynccnt 4" ::: "memory");
        } else {
            asm volatile("s_wait_asynccnt 0" ::: "memory");
        }
        __syncthreads();

#pragma unroll
        for (int kk = 0; kk < 2; ++kk) {
            const int k0 = s * 64 + kk * 32 + koff;
            FragAB a;
            a.u[0] = *(const uint4*)(arow + k0);
            a.u[1] = *(const uint4*)(arow + k0 + 16);
#pragma unroll
            for (int nt = 0; nt < 4; ++nt) {
                FragAB b;
                b.u[0] = *(const uint4*)&wbuf[buf][nt * 16 + row][kk * 32 + koff];
                b.u[1] = *(const uint4*)&wbuf[buf][nt * 16 + row][kk * 32 + koff + 16];
                acc[nt] = __builtin_amdgcn_wmma_f32_16x16x32_bf16(
                    false, a.v, false, b.v, (short)0, acc[nt], false, false);
            }
        }
        __syncthreads();  // all waves done reading before buf is overwritten
    }

#pragma unroll
    for (int nt = 0; nt < 4; ++nt) {
#pragma unroll
        for (int r = 0; r < 8; ++r) {
            const int m = m0 + r + half * 8;
            const int n = n0 + nt * 16 + row;
            const float v = acc[nt][r];
            if (MODE == 0)      Obf[(size_t)m * DMODEL + n] = f2bf(v);
            else if (MODE == 1) Obf[(size_t)n * T_SEQ + m]  = f2bf(v);
            else                Of[(size_t)m * DMODEL + n]  = v;
        }
    }
}

// Sliding-window attention core. One wave per (16-query tile, head).
// Q,K row-major bf16 [T][512]; Vt transposed bf16 [512][T]; O row-major bf16 [T][512].
__global__ void __launch_bounds__(128)
attn_kernel(const unsigned short* __restrict__ Q,
            const unsigned short* __restrict__ K,
            const unsigned short* __restrict__ Vt,
            unsigned short* __restrict__ O) {
    __shared__ __align__(16) unsigned short pbuf[4][16][96];

    const int lane = threadIdx.x & 31;
    const int wave = threadIdx.x >> 5;
    const int row  = lane & 15;
    const int half = lane >> 4;
    const int h    = blockIdx.y;
    const int t0   = (blockIdx.x * 4 + wave) * 16;
    const int dbase = h * DHEAD;
    const int koff  = half * 8;
    const int lanepart = half * 8 - row;  // (t - tk) lane-dependent term

    // Q A-fragments: two K=32 steps covering Dh=64
    FragAB qa[2];
    {
        const unsigned short* qrow = Q + (size_t)(t0 + row) * DMODEL + dbase;
#pragma unroll
        for (int ks = 0; ks < 2; ++ks) {
            const int k0 = ks * 32 + koff;
            qa[ks].u[0] = *(const uint4*)(qrow + k0);
            qa[ks].u[1] = *(const uint4*)(qrow + k0 + 16);
        }
    }

    // Scores over 6 key tiles spanning [t0-64, t0+32)
    float s[6][8];
#pragma unroll
    for (int j = 0; j < 6; ++j) {
        const int kt0 = t0 - 64 + j * 16;
        const bool tkok = (kt0 + row) >= 0;
        const int ktc = imin(imax(kt0 + row, 0), T_SEQ - 1);
        const unsigned short* krow = K + (size_t)ktc * DMODEL + dbase;
        f32x8 acc = {};
#pragma unroll
        for (int ks = 0; ks < 2; ++ks) {
            const int k0 = ks * 32 + koff;
            FragAB b;
            b.u[0] = *(const uint4*)(krow + k0);
            b.u[1] = *(const uint4*)(krow + k0 + 16);
            acc = __builtin_amdgcn_wmma_f32_16x16x32_bf16(
                false, qa[ks].v, false, b.v, (short)0, acc, false, false);
        }
#pragma unroll
        for (int r = 0; r < 8; ++r) {
            // t - tk = (64 - 16j + r) + lanepart; window-valid iff 0 <= t-tk < 64
            const unsigned diff = (unsigned)((64 - j * 16 + r) + lanepart);
            const bool valid = (diff < (unsigned)WIN) && tkok;
            s[j][r] = valid ? acc[r] * 0.125f : -3.0e38f;  // 1/sqrt(Dh)=1/8
        }
    }

    // Row-wise softmax; rows span 16 lanes of each half-wave -> xor-swizzle reduce
#pragma unroll
    for (int r = 0; r < 8; ++r) {
        float rm = s[0][r];
#pragma unroll
        for (int j = 1; j < 6; ++j) rm = fmaxf(rm, s[j][r]);
        rm = fmaxf(rm, swz_xor<8>(rm));
        rm = fmaxf(rm, swz_xor<4>(rm));
        rm = fmaxf(rm, swz_xor<2>(rm));
        rm = fmaxf(rm, swz_xor<1>(rm));
        float rs = 0.f;
#pragma unroll
        for (int j = 0; j < 6; ++j) { s[j][r] = __expf(s[j][r] - rm); rs += s[j][r]; }
        rs += swz_xor<8>(rs);
        rs += swz_xor<4>(rs);
        rs += swz_xor<2>(rs);
        rs += swz_xor<1>(rs);
        const float inv = 1.0f / rs;
#pragma unroll
        for (int j = 0; j < 6; ++j)
            pbuf[wave][r + half * 8][j * 16 + row] = f2bf(s[j][r] * inv);
    }
    __syncthreads();

    // P A-fragments from LDS: 3 K=32 steps over the 96-key span
    FragAB pa[3];
#pragma unroll
    for (int ks = 0; ks < 3; ++ks) {
        const int k0 = ks * 32 + koff;
        pa[ks].u[0] = *(const uint4*)&pbuf[wave][row][k0];
        pa[ks].u[1] = *(const uint4*)&pbuf[wave][row][k0 + 16];
    }

    // out = P @ V via Vt rows (contiguous along t); masked columns have p == 0
    const int tbase = t0 - 64;
#pragma unroll
    for (int nt = 0; nt < 4; ++nt) {
        f32x8 acc = {};
        const unsigned short* vrow = Vt + (size_t)(dbase + nt * 16 + row) * T_SEQ;
#pragma unroll
        for (int ks = 0; ks < 3; ++ks) {
            const int k0 = ks * 32 + koff;
            const int tt0 = imin(imax(tbase + k0, 0),      T_SEQ - 8);
            const int tt1 = imin(imax(tbase + k0 + 16, 0), T_SEQ - 8);
            FragAB b;
            b.u[0] = *(const uint4*)(vrow + tt0);
            b.u[1] = *(const uint4*)(vrow + tt1);
            acc = __builtin_amdgcn_wmma_f32_16x16x32_bf16(
                false, pa[ks].v, false, b.v, (short)0, acc, false, false);
        }
#pragma unroll
        for (int r = 0; r < 8; ++r)
            O[(size_t)(t0 + r + half * 8) * DMODEL + dbase + nt * 16 + row] = f2bf(acc[r]);
    }
}

extern "C" void kernel_launch(void* const* d_in, const int* in_sizes, int n_in,
                              void* d_out, int out_size, void* d_ws, size_t ws_size,
                              hipStream_t stream) {
    const float* x  = (const float*)d_in[0];
    const float* Wq = (const float*)d_in[1];
    const float* Wk = (const float*)d_in[2];
    const float* Wv = (const float*)d_in[3];
    const float* Wo = (const float*)d_in[4];
    // d_in[5] = window (always 64; baked in at compile time)

    const size_t NX = (size_t)T_SEQ * DMODEL;   // 2M elems
    const size_t NW = (size_t)DMODEL * DMODEL;  // 256K elems

    unsigned short* xb  = (unsigned short*)d_ws;
    unsigned short* qb  = xb  + NX;
    unsigned short* kb  = qb  + NX;
    unsigned short* vtb = kb  + NX;
    unsigned short* aob = vtb + NX;
    unsigned short* wqb = aob + NX;
    unsigned short* wkb = wqb + NW;
    unsigned short* wvb = wkb + NW;
    unsigned short* wob = wvb + NW;

    convert_f32_bf16<<<256, 256, 0, stream>>>(x,  xb,  (int)NX);
    convert_f32_bf16<<<64,  256, 0, stream>>>(Wq, wqb, (int)NW);
    convert_f32_bf16<<<64,  256, 0, stream>>>(Wk, wkb, (int)NW);
    convert_f32_bf16<<<64,  256, 0, stream>>>(Wv, wvb, (int)NW);
    convert_f32_bf16<<<64,  256, 0, stream>>>(Wo, wob, (int)NW);

    dim3 gg(T_SEQ / 64, DMODEL / 64);  // 64 x 8 blocks, 128 threads each
    gemm_x_wT<0><<<gg, 128, 0, stream>>>(xb, wqb, qb,  nullptr);   // Q row-major
    gemm_x_wT<0><<<gg, 128, 0, stream>>>(xb, wkb, kb,  nullptr);   // K row-major
    gemm_x_wT<1><<<gg, 128, 0, stream>>>(xb, wvb, vtb, nullptr);   // V transposed

    attn_kernel<<<dim3(T_SEQ / 64, NHEAD), 128, 0, stream>>>(qb, kb, vtb, aob);

    gemm_x_wT<2><<<gg, 128, 0, stream>>>(aob, wob, nullptr, (float*)d_out);
}
